// SingleHeadAttention_65884798321388
// MI455X (gfx1250) — matrix operand order repaired and checked by reading, not compiled, not run on hardware
//
#include <hip/hip_runtime.h>

// ---------------------------------------------------------------------------
// SingleHeadAttention (ReLU-attention, causal, row/ max(i,1) scaling)
// B=4, L=4096, D=256, fp32 in/out. bf16 WMMA (v_wmma_f32_16x16x32_bf16)
// with f32 accumulation for all four GEMM stages. K-chunk staging uses
// GLOBAL_LOAD_ASYNC_TO_LDS_B128 (ASYNCcnt) — CDNA5 async copy path.
// ---------------------------------------------------------------------------

typedef __attribute__((ext_vector_type(16))) __bf16          bf16x16;
typedef __attribute__((ext_vector_type(8)))  float           f32x8;
typedef __attribute__((ext_vector_type(8)))  unsigned short  u16x8;

union FragU {
    bf16x16 v;
    u16x8   h[2];
};
static_assert(sizeof(FragU) == 32, "frag size");

#define BATCH 4
#define LSEQ  4096
#define DIM   256

// ---------------------------------------------------------------------------
// fp32 -> bf16 conversion (grid-stride, native convert ops)
// ---------------------------------------------------------------------------
__global__ void cvt_f32_bf16(const float* __restrict__ src,
                             __bf16* __restrict__ dst, int n) {
    for (int i = blockIdx.x * blockDim.x + threadIdx.x; i < n;
         i += gridDim.x * blockDim.x)
        dst[i] = (__bf16)src[i];
}

// ---------------------------------------------------------------------------
// Y[m][n] = sum_k A[m][k] * W[n][k]   (i.e. Y = A @ W^T)
// A: [M x 256] bf16 row-major, W: [256 x 256] bf16 row-major.
// One wave computes a 16x64 strip (4 accumulators, A-fragment reused 4x).
// ---------------------------------------------------------------------------
__global__ __launch_bounds__(128) void proj_bf16(
    const unsigned short* __restrict__ A,
    const unsigned short* __restrict__ W,
    unsigned short* __restrict__ Y) {
    const int lane = threadIdx.x & 31;
    const int wave = threadIdx.x >> 5;
    const int lm = lane & 15, lh = lane >> 4;
    const int m0 = blockIdx.x * 16;
    const int nbase = wave * 64;

    const size_t arow = (size_t)(m0 + lm) * DIM;

    f32x8 acc[4];
    #pragma unroll
    for (int t = 0; t < 4; ++t) acc[t] = (f32x8){};

    #pragma unroll
    for (int k0 = 0; k0 < DIM; k0 += 32) {
        FragU a;
        a.h[0] = *(const u16x8*)(A + arow + k0 + lh * 16);       // K 0..7 / 16..23
        a.h[1] = *(const u16x8*)(A + arow + k0 + lh * 16 + 8);   // K 8..15 / 24..31
        #pragma unroll
        for (int t = 0; t < 4; ++t) {
            const size_t brow = (size_t)(nbase + t * 16 + lm) * DIM;
            FragU b;
            b.h[0] = *(const u16x8*)(W + brow + k0 + lh * 16);   // col=lm, K contiguous
            b.h[1] = *(const u16x8*)(W + brow + k0 + lh * 16 + 8);
            acc[t] = __builtin_amdgcn_wmma_f32_16x16x32_bf16(
                false, a.v, false, b.v, (short)0, acc[t], false, false);
        }
    }
    #pragma unroll
    for (int t = 0; t < 4; ++t)
        #pragma unroll
        for (int v = 0; v < 8; ++v)
            ((__bf16*)Y)[(size_t)(m0 + lh * 8 + v) * DIM + nbase + t * 16 + lm] =
                (__bf16)acc[t][v];
}

// Same GEMM but fp32 output (final ctx @ Wo^T -> d_out).
__global__ __launch_bounds__(128) void proj_f32out(
    const unsigned short* __restrict__ A,
    const unsigned short* __restrict__ W,
    float* __restrict__ Y) {
    const int lane = threadIdx.x & 31;
    const int wave = threadIdx.x >> 5;
    const int lm = lane & 15, lh = lane >> 4;
    const int m0 = blockIdx.x * 16;
    const int nbase = wave * 64;

    const size_t arow = (size_t)(m0 + lm) * DIM;

    f32x8 acc[4];
    #pragma unroll
    for (int t = 0; t < 4; ++t) acc[t] = (f32x8){};

    #pragma unroll
    for (int k0 = 0; k0 < DIM; k0 += 32) {
        FragU a;
        a.h[0] = *(const u16x8*)(A + arow + k0 + lh * 16);
        a.h[1] = *(const u16x8*)(A + arow + k0 + lh * 16 + 8);
        #pragma unroll
        for (int t = 0; t < 4; ++t) {
            const size_t brow = (size_t)(nbase + t * 16 + lm) * DIM;
            FragU b;
            b.h[0] = *(const u16x8*)(W + brow + k0 + lh * 16);
            b.h[1] = *(const u16x8*)(W + brow + k0 + lh * 16 + 8);
            acc[t] = __builtin_amdgcn_wmma_f32_16x16x32_bf16(
                false, a.v, false, b.v, (short)0, acc[t], false, false);
        }
    }
    #pragma unroll
    for (int t = 0; t < 4; ++t)
        #pragma unroll
        for (int v = 0; v < 8; ++v)
            Y[(size_t)(m0 + lh * 8 + v) * DIM + nbase + t * 16 + lm] = acc[t][v];
}

// ---------------------------------------------------------------------------
// Causal ReLU attention:
//   S = Q Kt ; S = relu(mask(S)) / max(row,1) ; ctx += S V
// Workgroup = 4 waves = 64 query rows of one batch. Key chunks of 32 staged
// in LDS: K via GLOBAL_LOAD_ASYNC_TO_LDS_B128 (row-major), V via VGPR
// transpose (Vch[dim][key]) so both WMMA B-operands read contiguous.
// ---------------------------------------------------------------------------
__global__ __launch_bounds__(128) void attn_kernel(
    const unsigned short* __restrict__ Qb,
    const unsigned short* __restrict__ Kb,
    const unsigned short* __restrict__ Vb,
    unsigned short* __restrict__ Cb) {
    __shared__ __align__(16) unsigned short Kch[32 * DIM];    // [key][dim]  16 KB
    __shared__ __align__(16) unsigned short Vch[DIM * 32];    // [dim][key]  16 KB
    __shared__ __align__(16) unsigned short Sbuf[4][16 * 32]; // per-wave S   4 KB

    const int tid  = threadIdx.x;
    const int lane = tid & 31;
    const int wave = tid >> 5;
    const int lm = lane & 15, lh = lane >> 4;
    const int rb = blockIdx.x;                 // 0..63 row block (64 rows)
    const int b  = blockIdx.y;                 // batch
    const int m0 = rb * 64 + wave * 16;        // first query row of this wave
    const size_t base = (size_t)b * LSEQ * DIM;

    f32x8 acc[16];
    #pragma unroll
    for (int i = 0; i < 16; ++i) acc[i] = (f32x8){};

    const int nchunks = 2 * rb + 2;            // keys 0 .. rb*64+63
    for (int c = 0; c < nchunks; ++c) {
        const int kbase = c * 32;
        __syncthreads();
        // --- stage K chunk [key][dim] with async global->LDS copies (16B/lane)
        for (int i = tid; i < 1024; i += 128) {
            int key = i >> 5;
            int d   = (i & 31) * 8;
            unsigned lds_off = (unsigned)(size_t)&Kch[(size_t)key * DIM + d];
            const unsigned short* gaddr =
                Kb + base + (size_t)(kbase + key) * DIM + d;
            asm volatile("global_load_async_to_lds_b128 %0, %1, off"
                         :: "v"(lds_off), "v"(gaddr) : "memory");
        }
        // --- stage V chunk transposed: Vch[dim][key]; key pairs -> b32 stores
        for (int i = tid; i < 512; i += 128) {
            int kp = (i >> 5) * 2;             // even key of the pair
            int d0 = (i & 31) * 8;
            u16x8 va = *(const u16x8*)(Vb + base + (size_t)(kbase + kp) * DIM + d0);
            u16x8 vb = *(const u16x8*)(Vb + base + (size_t)(kbase + kp + 1) * DIM + d0);
            #pragma unroll
            for (int e = 0; e < 8; ++e) {
                unsigned pk = (unsigned)va[e] | ((unsigned)vb[e] << 16);
                *(unsigned*)&Vch[(size_t)(d0 + e) * 32 + kp] = pk;
            }
        }
        asm volatile("s_wait_asynccnt 0" ::: "memory");
        __syncthreads();

        if (kbase <= m0 + 15) {                // uniform per wave: EXEC stays full
            if (c + 1 < nchunks)               // pull next K chunk toward L2
                __builtin_prefetch(Kb + base + (size_t)(kbase + 32) * DIM +
                                       (size_t)tid * 64, 0, 0);

            // ---- S(16x32) = Q_tile(16x256) @ K_chunk^T ----
            f32x8 s0 = {}, s1 = {};
            const size_t qrow = base + (size_t)(m0 + lm) * DIM;
            #pragma unroll
            for (int k0 = 0; k0 < DIM; k0 += 32) {
                FragU a, b0, b1;
                a.h[0]  = *(const u16x8*)(Qb + qrow + k0 + lh * 16);
                a.h[1]  = *(const u16x8*)(Qb + qrow + k0 + lh * 16 + 8);
                b0.h[0] = *(const u16x8*)&Kch[(size_t)lm * DIM + k0 + lh * 16];
                b0.h[1] = *(const u16x8*)&Kch[(size_t)lm * DIM + k0 + lh * 16 + 8];
                b1.h[0] = *(const u16x8*)&Kch[(size_t)(16 + lm) * DIM + k0 + lh * 16];
                b1.h[1] = *(const u16x8*)&Kch[(size_t)(16 + lm) * DIM + k0 + lh * 16 + 8];
                s0 = __builtin_amdgcn_wmma_f32_16x16x32_bf16(false, a.v, false, b0.v,
                                                             (short)0, s0, false, false);
                s1 = __builtin_amdgcn_wmma_f32_16x16x32_bf16(false, a.v, false, b1.v,
                                                             (short)0, s1, false, false);
            }
            // ---- causal mask + relu + 1/max(r,1); write A-layout bf16 tile ----
            __bf16* sw = (__bf16*)Sbuf[wave];
            #pragma unroll
            for (int v = 0; v < 8; ++v) {
                int m = lh * 8 + v;            // C layout: row = vgpr + 8*(lane>=16)
                int r = m0 + m;
                float scale = 1.0f / fmaxf((float)r, 1.0f);
                int c0 = kbase + lm;
                int c1 = kbase + 16 + lm;
                float x0 = (c0 <= r) ? fmaxf(s0[v], 0.0f) * scale : 0.0f;
                float x1 = (c1 <= r) ? fmaxf(s1[v], 0.0f) * scale : 0.0f;
                sw[m * 32 + lm]      = (__bf16)x0;
                sw[m * 32 + 16 + lm] = (__bf16)x1;
            }
            // wave-local cross-lane transpose through LDS: wait for DS stores
            asm volatile("s_wait_dscnt 0" ::: "memory");

            FragU sa;
            sa.h[0] = *(const u16x8*)&Sbuf[wave][lm * 32 + lh * 16];
            sa.h[1] = *(const u16x8*)&Sbuf[wave][lm * 32 + lh * 16 + 8];
            // ---- ctx(16x256) += S(16x32) @ V_chunk(32x256) ----
            #pragma unroll
            for (int nt = 0; nt < 16; ++nt) {
                FragU vb;
                vb.h[0] = *(const u16x8*)&Vch[(size_t)(nt * 16 + lm) * 32 + lh * 16];
                vb.h[1] = *(const u16x8*)&Vch[(size_t)(nt * 16 + lm) * 32 + lh * 16 + 8];
                acc[nt] = __builtin_amdgcn_wmma_f32_16x16x32_bf16(
                    false, sa.v, false, vb.v, (short)0, acc[nt], false, false);
            }
        }
    }
    // ---- write ctx as bf16 ----
    __bf16* co = (__bf16*)Cb;
    #pragma unroll
    for (int nt = 0; nt < 16; ++nt)
        #pragma unroll
        for (int v = 0; v < 8; ++v)
            co[base + (size_t)(m0 + lh * 8 + v) * DIM + nt * 16 + lm] =
                (__bf16)acc[nt][v];
}

// ---------------------------------------------------------------------------
// launch
// ---------------------------------------------------------------------------
extern "C" void kernel_launch(void* const* d_in, const int* in_sizes, int n_in,
                              void* d_out, int out_size, void* d_ws, size_t ws_size,
                              hipStream_t stream) {
    (void)in_sizes; (void)n_in; (void)out_size; (void)ws_size;
    const float* x  = (const float*)d_in[0];
    const float* Wq = (const float*)d_in[1];
    const float* Wk = (const float*)d_in[2];
    const float* Wv = (const float*)d_in[3];
    const float* Wo = (const float*)d_in[4];

    const int M  = BATCH * LSEQ;          // 16384 rows
    const int NX = M * DIM;               // 4,194,304 elements
    const size_t MB = 1024 * 1024;

    // workspace layout (bf16 as ushort); total ~42 MB
    unsigned char* ws = (unsigned char*)d_ws;
    unsigned short* xb  = (unsigned short*)(ws + 0 * MB);
    unsigned short* qb  = (unsigned short*)(ws + 8 * MB);
    unsigned short* kb  = (unsigned short*)(ws + 16 * MB);
    unsigned short* vb  = (unsigned short*)(ws + 24 * MB);
    unsigned short* cb  = (unsigned short*)(ws + 32 * MB);
    unsigned short* wqb = (unsigned short*)(ws + 40 * MB);
    unsigned short* wkb = wqb + DIM * DIM;
    unsigned short* wvb = wkb + DIM * DIM;
    unsigned short* wob = wvb + DIM * DIM;

    cvt_f32_bf16<<<1024, 256, 0, stream>>>(x, (__bf16*)xb, NX);
    cvt_f32_bf16<<<64, 256, 0, stream>>>(Wq, (__bf16*)wqb, DIM * DIM);
    cvt_f32_bf16<<<64, 256, 0, stream>>>(Wk, (__bf16*)wkb, DIM * DIM);
    cvt_f32_bf16<<<64, 256, 0, stream>>>(Wv, (__bf16*)wvb, DIM * DIM);
    cvt_f32_bf16<<<64, 256, 0, stream>>>(Wo, (__bf16*)wob, DIM * DIM);

    dim3 pg(M / 16);                      // wave per 16x64 strip, 4 waves/block
    proj_bf16<<<pg, 128, 0, stream>>>(xb, wqb, qb);
    proj_bf16<<<pg, 128, 0, stream>>>(xb, wkb, kb);
    proj_bf16<<<pg, 128, 0, stream>>>(xb, wvb, vb);

    attn_kernel<<<dim3(LSEQ / 64, BATCH), 128, 0, stream>>>(qb, kb, vb, cb);

    proj_f32out<<<pg, 128, 0, stream>>>(cb, wob, (float*)d_out);
}
